// MultiHeadAttention_45303315038955
// MI455X (gfx1250) — matrix operand order repaired and checked by reading, not compiled, not run on hardware
//
#include <hip/hip_runtime.h>

// ---------------------------------------------------------------------------
// CDNA5 (gfx1250) wave32 WMMA bf16 attention; TDM-staged flash attention.
// ---------------------------------------------------------------------------

typedef __attribute__((ext_vector_type(16))) __bf16 v16bf;
typedef __attribute__((ext_vector_type(8)))  float  v8f;
typedef __attribute__((ext_vector_type(4)))  unsigned v4u;
typedef __attribute__((ext_vector_type(8)))  int    v8i;
typedef __attribute__((ext_vector_type(4)))  int    v4i;

#define NEGBIG (-1e30f)

__device__ __forceinline__ __bf16 f2bf(float f) {   // RNE scalar convert
  unsigned u = __builtin_bit_cast(unsigned, f);
  u = (u + 0x7FFFu + ((u >> 16) & 1u)) >> 16;
  unsigned short s = (unsigned short)u;
  return __builtin_bit_cast(__bf16, s);
}

// pack two f32 into a dword of 2 bf16 (truncate) with one v_perm_b32
__device__ __forceinline__ unsigned pack2bf(float hi, float lo) {
  return __builtin_amdgcn_perm(__builtin_bit_cast(unsigned, hi),
                               __builtin_bit_cast(unsigned, lo), 0x07060302u);
}

union Frag16 { v16bf v; int4 q[2]; };

// Build a 16x32 16-bit WMMA operand fragment from two 16B runs.
__device__ __forceinline__ v16bf load_frag(const __bf16* p0, const __bf16* p1) {
  Frag16 f;
  f.q[0] = *(const int4*)p0;
  f.q[1] = *(const int4*)p1;
  return f.v;
}

// generic pointer to LDS byte offset (flat LDS aperture: offset in addr[31:0])
__device__ __forceinline__ unsigned lds_off(const void* p) {
  return (unsigned)(size_t)p;
}

// TDM: DMA a 2-D bf16 tile (tile_d0 contiguous elems x tile_d1 rows, row
// stride stride_elems) from global to contiguous LDS. D# per ISA 8.3-8.5.
// This toolchain exposes the 6-arg builtin (extra int32x8 group, zero-filled).
__device__ __forceinline__ void tdm_load_2d(unsigned ldsaddr, const void* gaddr,
                                            unsigned tile_d0, unsigned tile_d1,
                                            unsigned stride_elems) {
  size_t ga = (size_t)gaddr;
  v4u g0;
  g0[0] = 1u;                                   // count=1 (valid), user mode
  g0[1] = ldsaddr;                              // lds_addr
  g0[2] = (unsigned)ga;                         // global_addr[31:0]
  g0[3] = (unsigned)(ga >> 32) | (2u << 30);    // global_addr[56:32] | type=2
  v8i g1;
  g1[0] = (int)(1u << 16);                      // wg_mask=0, data_size=1 (2B)
  g1[1] = (int)((tile_d0 & 0xFFFFu) << 16);     // tensor_dim0[15:0]
  g1[2] = (int)((tile_d0 >> 16) | ((tile_d1 & 0xFFFFu) << 16)); // d0 hi|d1 lo
  g1[3] = (int)((tile_d1 >> 16) | ((tile_d0 & 0xFFFFu) << 16)); // d1 hi|tile_dim0
  g1[4] = (int)(tile_d1 & 0xFFFFu);             // tile_dim1 (tile_dim2=0)
  g1[5] = (int)stride_elems;                    // tensor_dim0_stride lo32
  g1[6] = 0;
  g1[7] = 0;
  v4i g2 = {0, 0, 0, 0};
  v4i g3 = {0, 0, 0, 0};
  v8i g4 = {0, 0, 0, 0, 0, 0, 0, 0};
  __builtin_amdgcn_tensor_load_to_lds(g0, g1, g2, g3, g4, 0);
}

// ---------------------------------------------------------------------------
// Prologue converters (memory-bound, run once per call)
// ---------------------------------------------------------------------------
__global__ void __launch_bounds__(256) cvt_f32_bf16_vec8(
    const float* __restrict__ in, __bf16* __restrict__ out, long n8) {
  long i = (long)blockIdx.x * 256 + threadIdx.x;
  if (i >= n8) return;
  const float4* p = (const float4*)in + 2 * i;
  float4 f0 = p[0], f1 = p[1];
  int4 o;
  o.x = (int)pack2bf(f0.y, f0.x);
  o.y = (int)pack2bf(f0.w, f0.z);
  o.z = (int)pack2bf(f1.y, f1.x);
  o.w = (int)pack2bf(f1.w, f1.z);
  ((int4*)out)[i] = o;
}

// W (KxN f32, row-major) -> Wt (NxK bf16, row-major)
__global__ void __launch_bounds__(256) cvt_w_transpose(
    const float* __restrict__ W, __bf16* __restrict__ Wt, int K, int N) {
  int k8s = K >> 3;
  int tid = blockIdx.x * 256 + threadIdx.x;
  if (tid >= N * k8s) return;
  int n  = tid / k8s;
  int k0 = (tid - n * k8s) << 3;
  float f[8];
#pragma unroll
  for (int i = 0; i < 8; ++i) f[i] = W[(size_t)(k0 + i) * N + n];
  int4 o;
  o.x = (int)pack2bf(f[1], f[0]);
  o.y = (int)pack2bf(f[3], f[2]);
  o.z = (int)pack2bf(f[5], f[4]);
  o.w = (int)pack2bf(f[7], f[6]);
  *(int4*)(Wt + (size_t)n * K + k0) = o;
}

// ---------------------------------------------------------------------------
// C = A(MxK bf16) * Bt^T (Bt is NxK bf16). One wave: 32x64 tile, K-step 32.
// ---------------------------------------------------------------------------
template <bool OUT_F32, bool TRANS_OUT>
__global__ void __launch_bounds__(256) gemm_bf16(
    const __bf16* __restrict__ A, const __bf16* __restrict__ Bt,
    void* __restrict__ Cout, int M, int N, int K, int T) {
  const int gwave = (blockIdx.x * blockDim.x + threadIdx.x) >> 5;
  const int lane  = threadIdx.x & 31;
  const int half  = lane >> 4;
  const int l16   = lane & 15;
  const int tiles_n = N >> 6;
  const int tm = gwave / tiles_n;
  const int tn = gwave - tm * tiles_n;
  if (tm * 32 >= M) return;
  const int row0 = tm << 5, col0 = tn << 6;

  v8f acc[2][4] = {};
  const __bf16* ar0 = A + (size_t)(row0 + l16) * K;
  const __bf16* ar1 = A + (size_t)(row0 + 16 + l16) * K;
  for (int k0 = 0; k0 < K; k0 += 32) {
    v16bf a0 = load_frag(ar0 + k0 + half * 16, ar0 + k0 + half * 16 + 8);
    v16bf a1 = load_frag(ar1 + k0 + half * 16, ar1 + k0 + half * 16 + 8);
#pragma unroll
    for (int nb = 0; nb < 4; ++nb) {
      const __bf16* bp = Bt + (size_t)(col0 + nb * 16 + l16) * K + k0 + half * 16;
      v16bf b = load_frag(bp, bp + 8);
      acc[0][nb] = __builtin_amdgcn_wmma_f32_16x16x32_bf16(
          false, a0, false, b, (short)0, acc[0][nb], false, false);
      acc[1][nb] = __builtin_amdgcn_wmma_f32_16x16x32_bf16(
          false, a1, false, b, (short)0, acc[1][nb], false, false);
    }
  }

  if constexpr (OUT_F32) {
    float* C = (float*)Cout;
#pragma unroll
    for (int mi = 0; mi < 2; ++mi)
#pragma unroll
      for (int nb = 0; nb < 4; ++nb)
#pragma unroll
        for (int r = 0; r < 8; ++r)
          C[(size_t)(row0 + mi * 16 + r + 8 * half) * N + col0 + nb * 16 + l16] =
              acc[mi][nb][r];
  } else if constexpr (TRANS_OUT) {
    __bf16* C = (__bf16*)Cout;
    const int bIdx = row0 / T;
    const int t0   = row0 - bIdx * T;
#pragma unroll
    for (int mi = 0; mi < 2; ++mi)
#pragma unroll
      for (int nb = 0; nb < 4; ++nb) {
        int n = col0 + nb * 16 + l16;
        __bf16* dst = C + ((size_t)(bIdx * N + n)) * T + (t0 + mi * 16 + 8 * half);
        int4 o;
        o.x = (int)pack2bf(acc[mi][nb][1], acc[mi][nb][0]);
        o.y = (int)pack2bf(acc[mi][nb][3], acc[mi][nb][2]);
        o.z = (int)pack2bf(acc[mi][nb][5], acc[mi][nb][4]);
        o.w = (int)pack2bf(acc[mi][nb][7], acc[mi][nb][6]);
        *(int4*)dst = o;
      }
  } else {
    __bf16* C = (__bf16*)Cout;
#pragma unroll
    for (int mi = 0; mi < 2; ++mi)
#pragma unroll
      for (int nb = 0; nb < 4; ++nb)
#pragma unroll
        for (int r = 0; r < 8; ++r)
          C[(size_t)(row0 + mi * 16 + r + 8 * half) * N + col0 + nb * 16 + l16] =
              f2bf(acc[mi][nb][r]);
  }
}

// ---------------------------------------------------------------------------
// RoPE in place on bf16 Q (BTx1024, 16 heads) / K (BTx256, 4 heads).
// ---------------------------------------------------------------------------
__global__ void __launch_bounds__(256) rope_bf16(
    __bf16* __restrict__ Qb, __bf16* __restrict__ Kb, int BT, int T) {
  int idx = blockIdx.x * 256 + threadIdx.x;
  if (idx >= BT * 20 * 32) return;
  int pair = idx & 31;
  int rest = idx >> 5;
  int head = rest % 20;
  int bt   = rest / 20;
  int t    = bt % T;
  float inv_freq = __powf(10000.0f, -(float)(2 * pair) * (1.0f / 64.0f));
  float s, c;
  __sincosf((float)t * inv_freq, &s, &c);
  unsigned* p = (head < 16)
                    ? (unsigned*)(Qb + (size_t)bt * 1024 + head * 64) + pair
                    : (unsigned*)(Kb + (size_t)bt * 256 + (head - 16) * 64) + pair;
  unsigned w = *p;
  float x1 = __builtin_bit_cast(float, w << 16);
  float x2 = __builtin_bit_cast(float, w & 0xffff0000u);
  float o1 = x1 * c - x2 * s;
  float o2 = x1 * s + x2 * c;
  *p = pack2bf(o2, o1);
}

// ---------------------------------------------------------------------------
// Flash attention. Block = 8 waves on one (b,h,128-query panel).
// Wave 0 streams K (32x64) and V^T (64x32) slabs into double-buffered LDS
// with TENSOR_LOAD_TO_LDS; prefetch of slab i+1 overlaps WMMA on slab i.
// ---------------------------------------------------------------------------
__global__ void __launch_bounds__(256) attn_fa_kernel(
    const __bf16* __restrict__ Qb, const __bf16* __restrict__ Kb,
    const __bf16* __restrict__ Vt, const int* __restrict__ amask,
    __bf16* __restrict__ Ob, int Bsz, int T) {
  __shared__ __bf16 ksh[2][32 * 64];   // keys x d          (2 x 4 KB)
  __shared__ __bf16 vsh[2][64 * 32];   // d x keys (V^T)    (2 x 4 KB)
  __shared__ __bf16 psh[8][16][32];    // per-wave P stage  (8 KB)

  const int lane = threadIdx.x & 31;
  const int wv   = threadIdx.x >> 5;
  const int half = lane >> 4;
  const int l16  = lane & 15;
  const int qtiles = T >> 7;
  const int qblk = blockIdx.x % qtiles;
  const int bh   = blockIdx.x / qtiles;
  const int h    = bh & 15;
  const int b    = bh >> 4;
  const int kvh  = h >> 2;
  const int q0   = qblk * 128 + wv * 16;
  const int kvmax = qblk * 128 + 128;        // block-uniform causal bound

  const __bf16* qp = Qb + ((size_t)(b * T + q0 + l16)) * 1024 + h * 64;
  v16bf qa0 = load_frag(qp + half * 16,      qp + half * 16 + 8);
  v16bf qa1 = load_frag(qp + 32 + half * 16, qp + 32 + half * 16 + 8);

  float mrow[8], lrow[8];
#pragma unroll
  for (int r = 0; r < 8; ++r) { mrow[r] = NEGBIG; lrow[r] = 0.f; }
  v8f o[4] = {};

  const __bf16* kbase = Kb + (size_t)b * T * 256 + kvh * 64;
  const __bf16* vbase = Vt + (size_t)(b * 256 + kvh * 64) * T;
  const int* mbase = amask + b * T;

  const int nslab = kvmax >> 5;
  if (wv == 0) {   // prologue: stage slab 0 into buffer 0
    tdm_load_2d(lds_off(&ksh[0][0]), kbase, 64, 32, 256);
    tdm_load_2d(lds_off(&vsh[0][0]), vbase, 32, 64, (unsigned)T);
  }

  for (int i = 0; i < nslab; ++i) {
    const int kv0 = i << 5;
    const int cur = i & 1;
    if (wv == 0) {
      if (i + 1 < nslab) {     // prefetch next slab into other buffer
        const int kn = kv0 + 32;
        tdm_load_2d(lds_off(&ksh[cur ^ 1][0]), kbase + (size_t)kn * 256, 64, 32, 256);
        tdm_load_2d(lds_off(&vsh[cur ^ 1][0]), vbase + kn, 32, 64, (unsigned)T);
        __builtin_amdgcn_s_wait_tensorcnt(2);  // current slab's 2 DMAs done
      } else {
        __builtin_amdgcn_s_wait_tensorcnt(0);
      }
    }
    __syncthreads();           // slab `cur` visible to all waves

    if (kv0 < q0 + 16) {       // wave-uniform causal skip
      v8f sa[2];
#pragma unroll
      for (int s = 0; s < 2; ++s) {
        const int key = kv0 + s * 16 + l16;          // always < kvmax <= T
        const __bf16* kp = &ksh[cur][(s * 16 + l16) * 64];
        v16bf kb0 = load_frag(kp + half * 16,      kp + half * 16 + 8);
        v16bf kb1 = load_frag(kp + 32 + half * 16, kp + 32 + half * 16 + 8);
        v8f acc = {};
        acc = __builtin_amdgcn_wmma_f32_16x16x32_bf16(false, qa0, false, kb0,
                                                      (short)0, acc, false, false);
        acc = __builtin_amdgcn_wmma_f32_16x16x32_bf16(false, qa1, false, kb1,
                                                      (short)0, acc, false, false);
        const int am = mbase[key];
#pragma unroll
        for (int r = 0; r < 8; ++r) {
          const int q = q0 + r + 8 * half;
          const bool ok = (key <= q) && (am != 0);
          sa[s][r] = ok ? acc[r] * 0.125f : NEGBIG;  // scale = 1/sqrt(64)
        }
      }
      // --- online softmax over this 32-key slab ---
#pragma unroll
      for (int r = 0; r < 8; ++r) {
        float v = fmaxf(sa[0][r], sa[1][r]);
#pragma unroll
        for (int off = 1; off < 16; off <<= 1) v = fmaxf(v, __shfl_xor(v, off, 32));
        const float mnew  = fmaxf(mrow[r], v);
        const float alpha = __expf(mrow[r] - mnew);
        float p0 = __expf(sa[0][r] - mnew);
        float p1 = __expf(sa[1][r] - mnew);
        sa[0][r] = p0; sa[1][r] = p1;
        float ps = p0 + p1;
#pragma unroll
        for (int off = 1; off < 16; off <<= 1) ps += __shfl_xor(ps, off, 32);
        lrow[r] = lrow[r] * alpha + ps;
        mrow[r] = mnew;
#pragma unroll
        for (int nb = 0; nb < 4; ++nb) o[nb][r] *= alpha;
      }
      // --- P: C-layout -> A-fragment via bf16 LDS slice (same-wave) ---
#pragma unroll
      for (int s = 0; s < 2; ++s)
#pragma unroll
        for (int r = 0; r < 8; ++r)
          psh[wv][r + 8 * half][s * 16 + l16] = f2bf(sa[s][r]);
      __asm__ volatile("s_wait_dscnt 0" ::: "memory");
      v16bf pa = load_frag(&psh[wv][l16][half * 16], &psh[wv][l16][half * 16 + 8]);
      // --- O += P * V from LDS (V^T rows: d, consecutive keys) ---
#pragma unroll
      for (int nb = 0; nb < 4; ++nb) {
        const __bf16* vp = &vsh[cur][(nb * 16 + l16) * 32];
        v16bf vb = load_frag(vp + half * 16, vp + half * 16 + 8);
        o[nb] = __builtin_amdgcn_wmma_f32_16x16x32_bf16(false, pa, false, vb,
                                                        (short)0, o[nb], false, false);
      }
    }
    __syncthreads();           // all waves done reading buffer `cur`
  }

  // epilogue: normalize, store bf16 (B,T,H*64) row-major
#pragma unroll
  for (int r = 0; r < 8; ++r) {
    const int q = q0 + r + 8 * half;
    const float inv = 1.0f / lrow[r];
#pragma unroll
    for (int nb = 0; nb < 4; ++nb)
      Ob[((size_t)(b * T + q)) * 1024 + h * 64 + nb * 16 + l16] =
          f2bf(o[nb][r] * inv);
  }
}

// ---------------------------------------------------------------------------
extern "C" void kernel_launch(void* const* d_in, const int* in_sizes, int n_in,
                              void* d_out, int out_size, void* d_ws, size_t ws_size,
                              hipStream_t stream) {
  const float* x  = (const float*)d_in[0];
  const int*   am = (const int*)d_in[1];
  const float* Wq = (const float*)d_in[2];
  const float* Wk = (const float*)d_in[3];
  const float* Wv = (const float*)d_in[4];
  const float* Wo = (const float*)d_in[5];

  const int T   = 2048;
  const int BT  = in_sizes[1];   // B*T
  const int Bsz = BT / T;

  // Workspace (bf16): x | Wq^T | Wk^T | Wv^T | Wo^T | Q | K | V^T | O  (~64 MB)
  __bf16* xbf = (__bf16*)d_ws;
  __bf16* Wqt = xbf + (size_t)BT * 1024;
  __bf16* Wkt = Wqt + (size_t)1024 * 1024;
  __bf16* Wvt = Wkt + (size_t)256 * 1024;
  __bf16* Wot = Wvt + (size_t)256 * 1024;
  __bf16* Qbf = Wot + (size_t)1024 * 1024;
  __bf16* Kbf = Qbf + (size_t)BT * 1024;
  __bf16* Vtt = Kbf + (size_t)BT * 256;
  __bf16* Obf = Vtt + (size_t)BT * 256;

  auto gblocks = [](int M, int N) {
    return ((M / 32) * (N / 64) * 32 + 255) / 256;
  };

  const long n8 = (long)BT * 1024 / 8;
  cvt_f32_bf16_vec8<<<(int)((n8 + 255) / 256), 256, 0, stream>>>(x, xbf, n8);
  cvt_w_transpose<<<(1024 * 128 + 255) / 256, 256, 0, stream>>>(Wq, Wqt, 1024, 1024);
  cvt_w_transpose<<<(256 * 128 + 255) / 256, 256, 0, stream>>>(Wk, Wkt, 1024, 256);
  cvt_w_transpose<<<(256 * 128 + 255) / 256, 256, 0, stream>>>(Wv, Wvt, 1024, 256);
  cvt_w_transpose<<<(1024 * 128 + 255) / 256, 256, 0, stream>>>(Wo, Wot, 1024, 1024);

  gemm_bf16<false, false><<<gblocks(BT, 1024), 256, 0, stream>>>(
      xbf, Wqt, Qbf, BT, 1024, 1024, T);
  gemm_bf16<false, false><<<gblocks(BT, 256), 256, 0, stream>>>(
      xbf, Wkt, Kbf, BT, 256, 1024, T);
  gemm_bf16<false, true><<<gblocks(BT, 256), 256, 0, stream>>>(
      xbf, Wvt, Vtt, BT, 256, 1024, T);

  const int rth = BT * 20 * 32;
  rope_bf16<<<(rth + 255) / 256, 256, 0, stream>>>(Qbf, Kbf, BT, T);

  attn_fa_kernel<<<Bsz * 16 * (T / 128), 256, 0, stream>>>(
      Qbf, Kbf, Vtt, am, Obf, Bsz, T);

  gemm_bf16<true, false><<<gblocks(BT, 1024), 256, 0, stream>>>(
      Obf, Wot, (float*)d_out, BT, 1024, 1024, T);
}
